// CHESHIRE_67224828117350
// MI455X (gfx1250) — compile-verified, matrix operand order checked by reading, not run on hardware
//
#include <hip/hip_runtime.h>
#include <hip/hip_bf16.h>

// MI455X / gfx1250 fused CHESHIRE link-prediction pipeline.
// Heavy math via v_wmma_f32_16x16x32_bf16; weights pre-swizzled into
// WMMA-fragment order so each B fragment is one aligned 32B vector load.
// Hyperedge feature gather uses gfx1250 async global->LDS copies (ASYNCcnt).

typedef __attribute__((ext_vector_type(16))) __bf16 v16bf;
typedef __attribute__((ext_vector_type(4)))  __bf16 v4bf;
typedef __attribute__((ext_vector_type(8)))  float  v8f;

#define NN 50000   // nodes
#define FF 512     // input features
#define DD 128     // embedding dim
#define CC 128     // conv channels
#define EE 20000   // hyperedges
// M = 8 nodes/hyperedge, cheb K = 3 (8-clique structure exploited analytically)

#define ENC_KT 16  // 512/32 K-tiles for encoder GEMM
#define CHB_KT 12  // 384/32 K-tiles for cheb GEMM

__device__ __forceinline__ __bf16 f2bf(float f) {
  unsigned u = __builtin_bit_cast(unsigned, f);
  u += 0x7FFFu + ((u >> 16) & 1u);            // round-to-nearest-even
  unsigned short h = (unsigned short)(u >> 16);
  return __builtin_bit_cast(__bf16, h);
}

// element e (0..15) of a 16-bit A/B WMMA fragment -> K offset, per ISA layout
__device__ __forceinline__ int kpat(int e, int hi) {
  return (e < 8) ? (hi * 8 + e) : (16 + hi * 8 + (e - 8));
}

// gfx1250 async global->LDS 16B copy, tracked by ASYNCcnt (ISA §15.18.3 op 98).
// ldsOff = offset within group segment (low 32 bits of flat shared pointer).
__device__ __forceinline__ void async_ld_b128(unsigned ldsOff,
                                              unsigned long long gaddr) {
  asm volatile("global_load_async_to_lds_b128 %0, %1, off"
               :: "v"(ldsOff), "v"(gaddr) : "memory");
}
__device__ __forceinline__ void wait_async0() {
  asm volatile("s_wait_asynccnt 0" ::: "memory");
}

// ---------------------------------------------------------------------------
// Kernel 1: convert weights to bf16 AND swizzle into fragment-native order:
//   frag index = (strip*nKt + kt)*32 + lane ; 16 contiguous bf16 per frag.
// ---------------------------------------------------------------------------
__global__ void CHESHIRE_prep_swizzle(const float* __restrict__ Wenc,
                                      const float* __restrict__ chebW,
                                      __bf16* __restrict__ WencSW,
                                      __bf16* __restrict__ chebSW) {
  int t = blockIdx.x * 256 + threadIdx.x;
  if (t < 8 * ENC_KT * 32) {                  // 4096 fragment-lanes
    int lane = t & 31, kt = (t >> 5) & (ENC_KT - 1), s = t >> 9;
    int hi = lane >> 4, lm = lane & 15;
    __bf16* dst = WencSW + t * 16;
#pragma unroll
    for (int e = 0; e < 16; ++e)
      dst[e] = f2bf(Wenc[(kt * 32 + kpat(e, hi)) * DD + s * 16 + lm]);
  }
  if (t < 8 * CHB_KT * 32) {                  // 3072 fragment-lanes
    int lane = t & 31, kt = (t >> 5) % CHB_KT, s = (t >> 5) / CHB_KT;
    int hi = lane >> 4, lm = lane & 15;
    __bf16* dst = chebSW + t * 16;
#pragma unroll
    for (int e = 0; e < 16; ++e)
      dst[e] = f2bf(chebW[(kt * 32 + kpat(e, hi)) * CC + s * 16 + lm]);
  }
}

// ---------------------------------------------------------------------------
// Kernel 2: x = clip(pos @ W_enc + b_enc, -1, 1)   [50000x512]@[512x128]
// 8 waves; wave w -> 16-col strip w, 4 row-tiles of 16; bf16 WMMA, K step 32.
// (f32 A read + in-kernel bf16 convert is roofline-optimal: a bf16 prep pass
//  would double the dominant 102MB HBM term.)
// ---------------------------------------------------------------------------
#define AST 520   // padded LDS row stride (bf16); 1040 B, 16B-aligned rows
__global__ __launch_bounds__(256)
void CHESHIRE_enc_gemm(const float* __restrict__ pos,
                       const v16bf* __restrict__ Bv,   // swizzled W_enc frags
                       const float* __restrict__ bias,
                       float* __restrict__ xout) {
  __shared__ alignas(16) __bf16 As[64 * AST];
  const int tid  = threadIdx.x;
  const int wave = tid >> 5, lane = tid & 31;
  const int lm   = lane & 15, hi = lane >> 4;
  const int wgRow = blockIdx.x * 64;

  // stage 64x512 A tile as bf16: float4 reads, 8B bf16 vector LDS stores
#pragma unroll 4
  for (int j = 0; j < 32; ++j) {
    int i  = j * 256 + tid;                   // float4 index in tile
    int r  = i >> 7;                          // 128 float4 per row
    int c4 = i & 127;
    int gr = wgRow + r; if (gr >= NN) gr = NN - 1;
    float4 f = ((const float4*)(pos + (long)gr * FF))[c4];
    v4bf p = { f2bf(f.x), f2bf(f.y), f2bf(f.z), f2bf(f.w) };
    *(v4bf*)(As + r * AST + c4 * 4) = p;
  }
  __syncthreads();

  v8f acc[4] = {};
#pragma unroll 2
  for (int kt = 0; kt < ENC_KT; ++kt) {
    int frag = (wave * ENC_KT + kt) * 32 + lane;
    v16bf b = Bv[frag];                       // one aligned 32B load
    __builtin_prefetch(Bv + frag + 32, 0, 3); // next K-tile, near scope
#pragma unroll
    for (int rt = 0; rt < 4; ++rt) {
      v16bf a;
#pragma unroll
      for (int e = 0; e < 16; ++e)
        a[e] = As[(rt * 16 + lm) * AST + kt * 32 + kpat(e, hi)];
      acc[rt] = __builtin_amdgcn_wmma_f32_16x16x32_bf16(
          false, a, false, b, (short)0, acc[rt], false, false);
    }
  }

  const int nb = wave * 16 + lm;              // output column for this lane
  const float bv = bias[nb];
#pragma unroll
  for (int rt = 0; rt < 4; ++rt) {
    int rowBase = wgRow + rt * 16;
    if (rowBase < NN) {                       // uniform guard (NN % 16 == 0)
#pragma unroll
      for (int v = 0; v < 8; ++v) {
        int row = rowBase + v + 8 * hi;       // C/D layout: M = v + 8*hi
        float h = fminf(1.f, fmaxf(-1.f, acc[rt][v] + bv));
        xout[(long)row * DD + nb] = h;
      }
    }
  }
}

// ---------------------------------------------------------------------------
// Kernel 3: fused GraphNorm + ChebConv + Hardtanh + pooling + linear+sigmoid.
// One WG = 2 hyperedges (16 rows). L̂h[i]=(h[i]-S)/7 (8-clique, deg=7), S1=-S0.
// GEMM [16x384]@[384x128]: 12 WMMA K-steps per wave, 8 waves = 8 col strips.
// C-layout puts one hyperedge's 8 rows in-lane -> register pooling.
// ---------------------------------------------------------------------------
#define TAST 392  // padded bf16 row stride; 784 B, 16B-aligned rows
__global__ __launch_bounds__(256)
void CHESHIRE_hyper_fused(const float* __restrict__ x,
                          const int* __restrict__ members,
                          const float* __restrict__ gnG,
                          const float* __restrict__ gnB,
                          const float* __restrict__ gnA,
                          const v16bf* __restrict__ Bv,  // swizzled cheb frags
                          const float* __restrict__ chebBias,
                          const float* __restrict__ linW,
                          const float* __restrict__ linB,
                          float* __restrict__ out) {
  __shared__ alignas(16) float  xs[16 * DD];
  __shared__ alignas(16) __bf16 Ta[16 * TAST];
  __shared__ float red[2];

  const int tid  = threadIdx.x;
  const int wave = tid >> 5, lane = tid & 31;
  const int lm   = lane & 15, hi = lane >> 4;
  const int wg   = blockIdx.x;                // hyperedges 2*wg, 2*wg+1

  if (tid < 2) red[tid] = 0.f;

  // gather 16 incidence rows of x: async global->LDS, 2x16B per thread,
  // no VGPR round-trip; completion tracked by ASYNCcnt.
  {
    int row  = tid >> 4;                      // 0..15
    int cb   = (tid & 15) * 32;               // byte offset within row
    int node = members[wg * 16 + row];
    unsigned long long g =
        (unsigned long long)(x + (long)node * DD) + (unsigned)cb;
    unsigned l = (unsigned)(size_t)(xs + row * DD) + (unsigned)cb;
    async_ld_b128(l, g);
    async_ld_b128(l + 16, g + 16);
  }
  wait_async0();                              // this wave's copies visible
  __syncthreads();                            // all waves' copies visible

  // GraphNorm + Chebyshev basis -> bf16 A matrix [16 x 384]
  {
    int g = tid >> 7;                         // group within WG (0..1)
    int c = tid & 127;                        // feature
    float v[8]; float mean = 0.f;
#pragma unroll
    for (int r = 0; r < 8; ++r) { v[r] = xs[(g * 8 + r) * DD + c]; mean += v[r]; }
    mean *= 0.125f;
    const float al = gnA[c], gm = gnG[c], bt = gnB[c];
    float var = 0.f;
#pragma unroll
    for (int r = 0; r < 8; ++r) { v[r] -= al * mean; var += v[r] * v[r]; }
    var *= 0.125f;
    const float inv = rsqrtf(var + 1e-5f);
    float t0[8]; float S0 = 0.f;
#pragma unroll
    for (int r = 0; r < 8; ++r) { t0[r] = gm * v[r] * inv + bt; S0 += t0[r]; }
#pragma unroll
    for (int r = 0; r < 8; ++r) {
      float t1 = (t0[r] - S0) * (1.f / 7.f);           // L̂ T0
      float t2 = (t1 + S0) * (2.f / 7.f) - t0[r];      // 2 L̂ T1 - T0 (S1=-S0)
      __bf16* rowp = Ta + (g * 8 + r) * TAST;
      rowp[c]          = f2bf(t0[r]);
      rowp[DD + c]     = f2bf(t1);
      rowp[2 * DD + c] = f2bf(t2);
    }
  }
  __syncthreads();

  // [16 x 384] @ [384 x 128]; wave -> 16-col strip; fully unrolled K loop
  v8f acc = {};
#pragma unroll
  for (int kt = 0; kt < CHB_KT; ++kt) {
    int frag = (wave * CHB_KT + kt) * 32 + lane;
    v16bf b = Bv[frag];                       // one aligned 32B load
    __builtin_prefetch(Bv + frag + 32, 0, 3);
    v16bf a;
#pragma unroll
    for (int e = 0; e < 16; ++e)
      a[e] = Ta[lm * TAST + kt * 32 + kpat(e, hi)];
    acc = __builtin_amdgcn_wmma_f32_16x16x32_bf16(
        false, a, false, b, (short)0, acc, false, false);
  }

  // Hardtanh + per-hyperedge pooling: this lane holds group `hi`'s 8 rows
  const int nb = wave * 16 + lm;
  const float cb = chebBias[nb];
  float mx = -2.f, mn = 2.f, ss = 0.f;
#pragma unroll
  for (int v = 0; v < 8; ++v) {
    float h = fminf(1.f, fmaxf(-1.f, acc[v] + cb));
    mx = fmaxf(mx, h); mn = fminf(mn, h); ss += h * h;
  }
  float part = (mx - mn) * linW[nb] + sqrtf(ss * 0.125f) * linW[CC + nb];
  atomicAdd(&red[hi], part);
  __syncthreads();

  if (tid < 2) {
    float z = red[tid] + linB[0];
    out[wg * 2 + tid] = 1.f / (1.f + __expf(-z));
  }
}

// ---------------------------------------------------------------------------
extern "C" void kernel_launch(void* const* d_in, const int* in_sizes, int n_in,
                              void* d_out, int out_size, void* d_ws, size_t ws_size,
                              hipStream_t stream) {
  const float* pos   = (const float*)d_in[0];
  const float* Wenc  = (const float*)d_in[1];
  const float* benc  = (const float*)d_in[2];
  const float* gnG   = (const float*)d_in[3];
  const float* gnB   = (const float*)d_in[4];
  const float* gnA   = (const float*)d_in[5];
  const float* chebW = (const float*)d_in[6];
  const float* chebb = (const float*)d_in[7];
  const float* linW  = (const float*)d_in[8];
  const float* linb  = (const float*)d_in[9];
  const int*   mem   = (const int*)d_in[10];
  // d_in[11] (edge_index) / d_in[12] (batch) are structurally determined and unused.
  float* out = (float*)d_out;

  char* ws = (char*)d_ws;
  __bf16* WencSW = (__bf16*)ws;                      // 4096 frags * 32B (128 KB)
  __bf16* chebSW = (__bf16*)(ws + 131072);           // 3072 frags * 32B ( 96 KB)
  float*  xbuf   = (float*)(ws + 131072 + 98304);    // 50000*128 f32  (25.6 MB)

  CHESHIRE_prep_swizzle<<<16, 256, 0, stream>>>(Wenc, chebW, WencSW, chebSW);
  CHESHIRE_enc_gemm<<<(NN + 63) / 64, 256, 0, stream>>>(
      pos, (const v16bf*)WencSW, benc, xbuf);
  CHESHIRE_hyper_fused<<<EE / 2, 256, 0, stream>>>(
      xbuf, mem, gnG, gnB, gnA, (const v16bf*)chebSW, chebb, linW, linb, out);
}
